// LlamaTernaryAttention_86311662781119
// MI455X (gfx1250) — compile-verified, matrix operand order checked by reading, not run on hardware
//
#include <hip/hip_runtime.h>
#include <hip/hip_bf16.h>

typedef __bf16 bf16_t;
typedef __attribute__((ext_vector_type(16))) __bf16 v16bf;
typedef __attribute__((ext_vector_type(8)))  __bf16 v8bf;
typedef __attribute__((ext_vector_type(4)))  __bf16 v4bf;
typedef __attribute__((ext_vector_type(8)))  float  v8f;
typedef __attribute__((ext_vector_type(4)))  float  v4f;

#define WMMA_BF16(a, b, c) \
    __builtin_amdgcn_wmma_f32_16x16x32_bf16(false, (a), false, (b), (short)0, (c), false, false)

__device__ __forceinline__ v8f zero8() {
    v8f z = {0.f, 0.f, 0.f, 0.f, 0.f, 0.f, 0.f, 0.f};
    return z;
}

// Low 32 bits of a generic pointer into LDS == LDS byte offset (flat->LDS truncation).
__device__ __forceinline__ uint32_t lds_addr32(const void* p) {
    return (uint32_t)(uintptr_t)p;
}

// CDNA5 async DMA: global -> LDS, 16 bytes per lane, GVS addressing
// (saddr = 64-bit SGPR base, vaddr = 32-bit per-lane byte offset,
//  vdst = per-lane LDS byte address). Tracked by ASYNCcnt.
__device__ __forceinline__ void async_load_b128(uint32_t lds_addr, uint32_t goff,
                                                const void* sbase) {
    asm volatile("global_load_async_to_lds_b128 %0, %1, %2"
                 :
                 : "v"(lds_addr), "v"(goff), "s"(sbase)
                 : "memory");
}

__device__ __forceinline__ void wait_asynccnt0() {
    asm volatile("s_wait_asynccnt 0" ::: "memory");
}

// Load one 16x32-bf16 WMMA operand fragment from an LDS row.
// ISA layout (16-bit A 16x32): lanes 0-15 hold K={0..7,16..23}, lanes 16-31 hold K={8..15,24..31}.
__device__ __forceinline__ v16bf load_frag(const bf16_t* base, int half) {
    v16bf f;
    *(v8bf*)&f       = *(const v8bf*)(base + half * 8);
    *((v8bf*)&f + 1) = *(const v8bf*)(base + 16 + half * 8);
    return f;
}

// ---------------------------------------------------------------------------
// GEMM: C[M][N] = A[M][K] @ W[N][K]^T   (fp32 or bf16 A; fp32 W; fp32 or bf16 C)
// Block tile 128x128, K-step 32, 256 threads = 8 waves, each wave 64x32.
// bf16 A tiles are staged with global_load_async_to_lds_b128.
// ---------------------------------------------------------------------------
#define BM 128
#define BN 128
#define BK 32
#define LDK 40  // padded LDS row stride (bf16): 80 B, multiple of 16 B

template <bool A_BF16, bool OUT_BF16>
__global__ __launch_bounds__(256) void gemm_wmma_kernel(const void* __restrict__ Ap,
                                                        const float* __restrict__ W,
                                                        void* __restrict__ Cp,
                                                        int N, int K) {
    __shared__ bf16_t As[BM * LDK];
    __shared__ bf16_t Bs[BN * LDK];

    const int tid  = threadIdx.x;
    const int lane = tid & 31;
    const int wave = tid >> 5;
    const int wm   = wave >> 2;   // 0..1
    const int wn   = wave & 3;    // 0..3
    const int half = lane >> 4;
    const int lrow = lane & 15;
    const long row0 = (long)blockIdx.y * BM;
    const long col0 = (long)blockIdx.x * BN;

    v8f acc[4][2];
#pragma unroll
    for (int t = 0; t < 4; ++t)
#pragma unroll
        for (int u = 0; u < 2; ++u) acc[t][u] = zero8();

    for (int kt = 0; kt < K; kt += BK) {
        // ---- stage A tile ----
        if constexpr (A_BF16) {
            // already bf16: async DMA straight into LDS (512 16-byte slots)
            const bf16_t* abase = (const bf16_t*)Ap + row0 * (long)K + kt;
#pragma unroll
            for (int i = 0; i < 2; ++i) {
                int s = tid + i * 256;
                int r = s >> 2, c = (s & 3) * 8;
                async_load_b128(lds_addr32(As + r * LDK + c),
                                (uint32_t)((r * (long)K + c) * sizeof(bf16_t)), abase);
            }
        }
#pragma unroll
        for (int i = 0; i < 4; ++i) {
            int s = tid + i * 256;       // 1024 4-element slots
            int r = s >> 3;              // 0..127
            int c = (s & 7) * 4;         // 0..28
            if constexpr (!A_BF16) {
                const float* A = (const float*)Ap;
                v4f fv = *(const v4f*)(A + (row0 + r) * (long)K + kt + c);
                v4bf av;
                av[0] = (bf16_t)fv[0]; av[1] = (bf16_t)fv[1];
                av[2] = (bf16_t)fv[2]; av[3] = (bf16_t)fv[3];
                *(v4bf*)(As + r * LDK + c) = av;
            }
            v4f wv = *(const v4f*)(W + (col0 + r) * (long)K + kt + c);
            v4bf wb;
            wb[0] = (bf16_t)wv[0]; wb[1] = (bf16_t)wv[1];
            wb[2] = (bf16_t)wv[2]; wb[3] = (bf16_t)wv[3];
            *(v4bf*)(Bs + r * LDK + c) = wb;
        }
        // prefetch next K-step tile rows (global_prefetch_b8)
        if (kt + BK < K) {
            int r = tid >> 1, c = (tid & 1) * 16;
            __builtin_prefetch(W + (col0 + r) * (long)K + kt + BK + c, 0, 0);
            if constexpr (!A_BF16)
                __builtin_prefetch((const float*)Ap + (row0 + r) * (long)K + kt + BK + c, 0, 0);
        }
        if constexpr (A_BF16) wait_asynccnt0();
        __syncthreads();

        // ---- WMMA: 4 M-tiles x 2 N-tiles per wave ----
        v16bf af[4], bf[2];
#pragma unroll
        for (int t = 0; t < 4; ++t)
            af[t] = load_frag(As + (wm * 64 + t * 16 + lrow) * LDK, half);
#pragma unroll
        for (int u = 0; u < 2; ++u)
            bf[u] = load_frag(Bs + (wn * 32 + u * 16 + lrow) * LDK, half);
#pragma unroll
        for (int t = 0; t < 4; ++t)
#pragma unroll
            for (int u = 0; u < 2; ++u) acc[t][u] = WMMA_BF16(af[t], bf[u], acc[t][u]);
        __syncthreads();
    }

    // ---- epilogue: C-tile layout -> row m = e + 8*half, col n = lane&15 ----
#pragma unroll
    for (int t = 0; t < 4; ++t)
#pragma unroll
        for (int u = 0; u < 2; ++u)
#pragma unroll
            for (int e = 0; e < 8; ++e) {
                long m = row0 + wm * 64 + t * 16 + half * 8 + e;
                long n = col0 + wn * 32 + u * 16 + lrow;
                if constexpr (OUT_BF16)
                    ((bf16_t*)Cp)[m * N + n] = (bf16_t)acc[t][u][e];
                else
                    ((float*)Cp)[m * N + n] = acc[t][u][e];
            }
}

// ---------------------------------------------------------------------------
// RoPE (in-place on bf16 activations laid out [S][heads*128])
// ---------------------------------------------------------------------------
__global__ __launch_bounds__(256) void rope_kernel(bf16_t* __restrict__ x,
                                                   const int* __restrict__ pos_ids,
                                                   int n_heads, int S) {
    int idx = blockIdx.x * blockDim.x + threadIdx.x;  // (s, head, i<64)
    int i  = idx & 63;
    int t  = idx >> 6;
    int hh = t % n_heads;
    int s  = t / n_heads;
    if (s >= S) return;
    float pos = (float)pos_ids[s];
    float inv = __powf(10000.0f, -(float)i * (1.0f / 64.0f));
    float ang = pos * inv;
    float sn, cs;
    __sincosf(ang, &sn, &cs);
    bf16_t* base = x + (long)s * (n_heads * 128) + hh * 128 + i;
    float x1 = (float)base[0];
    float x2 = (float)base[64];
    base[0]  = (bf16_t)(x1 * cs - x2 * sn);
    base[64] = (bf16_t)(x2 * cs + x1 * sn);
}

// ---------------------------------------------------------------------------
// Flash attention, causal, GQA (32 Q heads, 8 KV heads, d=128).
// One block = (head, 128-query tile). 8 waves, each owns 16 query rows.
// Q/K tiles staged via async DMA to LDS; V staged through VGPRs (transpose).
// ---------------------------------------------------------------------------
#define QSTR 136  // padded LDS row stride (bf16): 272 B, multiple of 16 B

__global__ __launch_bounds__(256) void flash_attn_kernel(const bf16_t* __restrict__ Q,
                                                         const bf16_t* __restrict__ Kb,
                                                         const bf16_t* __restrict__ Vb,
                                                         bf16_t* __restrict__ O) {
    extern __shared__ bf16_t smem[];
    bf16_t* Qs = smem;                 // [128][QSTR]  Q tile, row-major (q, dim)
    bf16_t* Ks = Qs + 128 * QSTR;      // [128][QSTR]  K tile, row-major (key, dim)
    bf16_t* Vt = Ks + 128 * QSTR;      // [128][QSTR]  V tile, transposed (dim, key)
    bf16_t* Ps = Vt + 128 * QSTR;      // [8*16][QSTR] per-wave P scratch

    const int tid  = threadIdx.x;
    const int lane = tid & 31;
    const int wave = tid >> 5;
    const int half = lane >> 4;
    const int lrow = lane & 15;
    const int qt  = blockIdx.x;
    const int h   = blockIdx.y;
    const int kvh = h >> 2;            // GQA: 4 Q heads per KV head
    const int q0  = qt * 128;
    const long qld = 32 * 128;
    const long kld = 8 * 128;
    const float scale = 0.08838834764831845f;  // 1/sqrt(128)

    // ---- async-stage Q tile (bf16, layout-preserving) ----
    {
        const bf16_t* qbase = Q + (long)q0 * qld + h * 128;
#pragma unroll
        for (int i = 0; i < 8; ++i) {
            int s = tid + i * 256;     // 2048 16-byte slots
            int r = s >> 4, c = (s & 15) * 8;
            async_load_b128(lds_addr32(Qs + r * QSTR + c),
                            (uint32_t)((r * qld + c) * sizeof(bf16_t)), qbase);
        }
    }

    v8f oacc[8];
    float msc[8], lsum[8];
#pragma unroll
    for (int d = 0; d < 8; ++d) oacc[d] = zero8();
#pragma unroll
    for (int e = 0; e < 8; ++e) { msc[e] = -3.0e38f; lsum[e] = 0.f; }

    for (int kt = 0; kt <= qt; ++kt) {
        const int k0 = kt * 128;
        __syncthreads();  // prior-iteration readers done before restaging K/V

        // ---- async-stage K tile; stage V transposed through VGPRs ----
        const bf16_t* kbase = Kb + (long)k0 * kld + kvh * 128;
#pragma unroll
        for (int i = 0; i < 8; ++i) {
            int s = tid + i * 256;
            int r = s >> 4, c = (s & 15) * 8;
            async_load_b128(lds_addr32(Ks + r * QSTR + c),
                            (uint32_t)((r * kld + c) * sizeof(bf16_t)), kbase);
            v8bf vv = *(const v8bf*)(Vb + (long)(k0 + r) * kld + kvh * 128 + c);
#pragma unroll
            for (int j = 0; j < 8; ++j) Vt[(c + j) * QSTR + r] = vv[j];
        }
        wait_asynccnt0();   // Q (first iter) + K async writes visible
        __syncthreads();

        // ---- S = Q K^T : 16 q-rows x 128 keys per wave ----
        v8f sc[8];
#pragma unroll
        for (int n = 0; n < 8; ++n) sc[n] = zero8();
#pragma unroll
        for (int kk = 0; kk < 4; ++kk) {
            v16bf a = load_frag(Qs + (wave * 16 + lrow) * QSTR + kk * 32, half);
#pragma unroll
            for (int n = 0; n < 8; ++n) {
                v16bf b = load_frag(Ks + (n * 16 + lrow) * QSTR + kk * 32, half);
                sc[n] = WMMA_BF16(a, b, sc[n]);
            }
        }

        // ---- causal mask + online softmax (rows live in elements, cols in lanes) ----
        bf16_t* Pw = Ps + wave * 16 * QSTR;
        const int qrow_base = q0 + wave * 16 + half * 8;
#pragma unroll
        for (int e = 0; e < 8; ++e) {
            const int qg = qrow_base + e;
            float sv[8];
            float rmax = -3.0e38f;
#pragma unroll
            for (int n = 0; n < 8; ++n) {
                int kg = k0 + n * 16 + lrow;
                float v = sc[n][e] * scale;
                v = (kg <= qg) ? v : -3.0e38f;
                sv[n] = v;
                rmax = fmaxf(rmax, v);
            }
#pragma unroll
            for (int m = 8; m >= 1; m >>= 1) rmax = fmaxf(rmax, __shfl_xor(rmax, m, 32));
            float newm = fmaxf(msc[e], rmax);
            float corr = __expf(msc[e] - newm);
            float psum = 0.f;
            bf16_t* prow = Pw + (half * 8 + e) * QSTR;
#pragma unroll
            for (int n = 0; n < 8; ++n) {
                float p = __expf(sv[n] - newm);
                psum += p;
                prow[n * 16 + lrow] = (bf16_t)p;
            }
#pragma unroll
            for (int m = 8; m >= 1; m >>= 1) psum += __shfl_xor(psum, m, 32);
            lsum[e] = lsum[e] * corr + psum;
            msc[e]  = newm;
#pragma unroll
            for (int d = 0; d < 8; ++d) oacc[d][e] = oacc[d][e] * corr;
        }

        // ---- O += P V : contraction over 128 keys ----
#pragma unroll
        for (int kk = 0; kk < 4; ++kk) {
            v16bf a = load_frag(Pw + lrow * QSTR + kk * 32, half);
#pragma unroll
            for (int d = 0; d < 8; ++d) {
                v16bf b = load_frag(Vt + (d * 16 + lrow) * QSTR + kk * 32, half);
                oacc[d] = WMMA_BF16(a, b, oacc[d]);
            }
        }
    }

    // ---- epilogue: normalize and store bf16 ----
#pragma unroll
    for (int d = 0; d < 8; ++d)
#pragma unroll
        for (int e = 0; e < 8; ++e) {
            long q = q0 + wave * 16 + half * 8 + e;
            O[q * qld + h * 128 + d * 16 + lrow] = (bf16_t)(oacc[d][e] / lsum[e]);
        }
}

// ---------------------------------------------------------------------------
extern "C" void kernel_launch(void* const* d_in, const int* in_sizes, int n_in,
                              void* d_out, int out_size, void* d_ws, size_t ws_size,
                              hipStream_t stream) {
    const float* hidden = (const float*)d_in[0];
    const float* Wq     = (const float*)d_in[1];
    const float* Wk     = (const float*)d_in[2];
    const float* Wv     = (const float*)d_in[3];
    const float* Wo     = (const float*)d_in[4];
    const int*   pos    = (const int*)d_in[5];
    float* out = (float*)d_out;

    const int S = 2048, H = 4096, QD = 4096, KVD = 1024;

    bf16_t* qb = (bf16_t*)d_ws;               // [S][4096]
    bf16_t* kb = qb + (size_t)S * QD;         // [S][1024]
    bf16_t* vb = kb + (size_t)S * KVD;        // [S][1024]
    bf16_t* ab = vb + (size_t)S * KVD;        // [S][4096] attention output

    dim3 blk(256);

    // Q/K/V projections (fp32 in -> bf16 out)
    gemm_wmma_kernel<false, true><<<dim3(QD / BN, S / BM), blk, 0, stream>>>(hidden, Wq, qb, QD, H);
    gemm_wmma_kernel<false, true><<<dim3(KVD / BN, S / BM), blk, 0, stream>>>(hidden, Wk, kb, KVD, H);
    gemm_wmma_kernel<false, true><<<dim3(KVD / BN, S / BM), blk, 0, stream>>>(hidden, Wv, vb, KVD, H);

    // RoPE on Q and K
    int nq = S * 32 * 64, nk = S * 8 * 64;
    rope_kernel<<<(nq + 255) / 256, blk, 0, stream>>>(qb, pos, 32, S);
    rope_kernel<<<(nk + 255) / 256, blk, 0, stream>>>(kb, pos, 8, S);

    // Flash attention: grid (q-tiles, heads)
    size_t smem = (size_t)(4 * 128) * QSTR * sizeof(bf16_t);
    flash_attn_kernel<<<dim3(S / 128, 32), blk, smem, stream>>>(qb, kb, vb, ab);

    // Output projection (bf16 in -> fp32 out)
    gemm_wmma_kernel<true, false><<<dim3(H / BN, S / BM), blk, 0, stream>>>(ab, Wo, out, H, QD);
}
